// CCN_Layer_75307956568415
// MI455X (gfx1250) — compile-verified, matrix operand order checked by reading, not run on hardware
//
#include <hip/hip_runtime.h>
#include <hip/hip_bf16.h>

typedef __attribute__((ext_vector_type(16))) _Float16 v16h;
typedef __attribute__((ext_vector_type(8)))  float    v8f;

#define DD 16     // receptive field size
#define FF 3      // feature dim
#define WAVES_PER_BLOCK 8

// One wave (32 lanes) per node. Per child c: one v_wmma_f32_16x16x32_f16
// computes outer(chi, chi) (chi in K=0 slot, zeros elsewhere), then the
// 16x16 f32 accumulator is scaled by the 3 feature scalars and streamed
// out with non-temporal stores (output ~984 MB >> 192 MB L2).
__global__ __launch_bounds__(256) void ccn_promote_kernel(
    const float* __restrict__ tensors,   // [N, 3]
    const int*   __restrict__ neigh,     // [N, 16]
    float*       __restrict__ out,       // promotions [N,16,16,16,3] ++ new_parts [N,16]
    int N)
{
  __shared__ int vals[WAVES_PER_BLOCK][DD];     // original neighbor order
  __shared__ int sortedv[WAVES_PER_BLOCK][DD];  // sorted neighbor ids

  const int tid  = threadIdx.x;
  const int wave = tid >> 5;
  const int lane = tid & 31;
  const int node = blockIdx.x * WAVES_PER_BLOCK + wave;
  const bool valid = (node < N);

  // --- load neighbor row (lanes 0..15) ---
  if (valid && lane < DD)
    vals[wave][lane] = neigh[node * DD + lane];
  __syncthreads();

  // --- stable rank sort of the 16 ids (reference: new_parts = sort(neigh)) ---
  if (valid && lane < DD) {
    const int v = vals[wave][lane];
    int rank = 0;
#pragma unroll
    for (int j = 0; j < DD; ++j) {
      const int u = vals[wave][j];
      rank += (u < v) || (u == v && j < lane);
    }
    sortedv[wave][rank] = v;
  }
  __syncthreads();

  if (!valid) return;  // wave-uniform exit after all barriers

  const int half = lane >> 4;              // 0: lanes 0-15, 1: lanes 16-31
  const int b    = lane & 15;              // D-matrix column for this lane
  const int myS  = sortedv[wave][b];       // sorted id for this lane's slot

  float* __restrict__ outProm  = out;
  float* __restrict__ outParts = out + (size_t)N * DD * DD * DD * FF;

  // --- new_parts output (ids are < 2^24, exact in f32) ---
  if (lane < DD)
    outParts[(size_t)node * DD + lane] = (float)sortedv[wave][lane];

  const size_t nodeBase = (size_t)node * (DD * DD * DD * FF);

  for (int c = 0; c < DD; ++c) {
    const int nc = __builtin_amdgcn_readfirstlane(vals[wave][c]);

    // feature scalars of child c (uniform across wave; tensors is L2-resident)
    const float f0 = tensors[nc * FF + 0];
    const float f1 = tensors[nc * FF + 1];
    const float f2 = tensors[nc * FF + 2];

    // chi occupies K=0 of A (rows) and K=0 of B (cols): element 0 of
    // lanes 0..15 per the 16-bit A(16x32)/B(32x16) layouts; all else zero.
    const _Float16 chi =
        (half == 0 && myS == nc) ? (_Float16)1.0f : (_Float16)0.0f;
    v16h ab = {};
    ab[0] = chi;

    v8f acc = {};
    acc = __builtin_amdgcn_wmma_f32_16x16x32_f16(
        /*neg_a=*/false, ab, /*neg_b=*/false, ab,
        /*c_mod=*/(short)0, acc, /*reuse_a=*/false, /*reuse_b=*/false);

    const size_t cBase = nodeBase + (size_t)c * (DD * DD * FF);
#pragma unroll
    for (int v = 0; v < 8; ++v) {
      const int   a  = v + 8 * half;             // C/D layout: row = v + 8*(lane>=16)
      const float dv = acc[v];
      const size_t o = cBase + (size_t)(a * DD + b) * FF;
      __builtin_nontemporal_store(dv * f0, outProm + o + 0);
      __builtin_nontemporal_store(dv * f1, outProm + o + 1);
      __builtin_nontemporal_store(dv * f2, outProm + o + 2);
    }
  }
}

extern "C" void kernel_launch(void* const* d_in, const int* in_sizes, int n_in,
                              void* d_out, int out_size, void* d_ws, size_t ws_size,
                              hipStream_t stream) {
  const float* tensors = (const float*)d_in[0];   // [N, 3] float32
  const int*   neigh   = (const int*)d_in[1];     // [N, 16] int32
  float*       out     = (float*)d_out;

  const int N = in_sizes[1] / DD;                 // 20000
  const int nodesPerBlock = WAVES_PER_BLOCK;      // 8 waves of 32 = 256 threads
  const int grid = (N + nodesPerBlock - 1) / nodesPerBlock;

  ccn_promote_kernel<<<grid, 256, 0, stream>>>(tensors, neigh, out, N);
}
